// MultiHeadAttention_31679678776005
// MI455X (gfx1250) — matrix operand order repaired and checked by reading, not compiled, not run on hardware
//
#include <hip/hip_runtime.h>
#include <hip/hip_bf16.h>

typedef __attribute__((ext_vector_type(16))) __bf16 bf16x16;
typedef __attribute__((ext_vector_type(8)))  __bf16 bf16x8;
typedef __attribute__((ext_vector_type(4)))  __bf16 bf16x4;
typedef __attribute__((ext_vector_type(8)))  float  f32x8;
typedef __attribute__((ext_vector_type(4)))  unsigned int u32x4;
typedef __attribute__((ext_vector_type(8)))  int i32x8;
typedef __attribute__((ext_vector_type(4)))  int i32x4;

#define D_MODEL 1024
#define NH      16
#define HD      64
#define SEQ     2048
#define BATCH   4
#define ROWS    (BATCH * SEQ)   // 8192

// ---------- WMMA fragment helpers (contiguous k = 16*g + j order, same for A and B) ----------

// A (16x32 bf16, row-major source): lane l = g*16+m holds row m,
//  elems j -> K = g*16+j  => one contiguous 32B load.
static __device__ __forceinline__ bf16x16 load_a_frag(const __bf16* p, int ld) {
  int l = threadIdx.x & 31;
  int g = l >> 4, m = l & 15;
  return *(const bf16x16*)(p + (size_t)m * ld + g * 16);
}

// B (32x16 bf16) from an N-major (transposed) source: lane l = g*16+n holds
//  column n, elems j -> K = g*16+j  => one contiguous 32B load.
static __device__ __forceinline__ bf16x16 load_b_frag(const __bf16* p, int ld) {
  int l = threadIdx.x & 31;
  int g = l >> 4, n = l & 15;
  return *(const bf16x16*)(p + (size_t)n * ld + g * 16);
}

static __device__ __forceinline__ f32x8 wmma_bf16(bf16x16 a, bf16x16 b, f32x8 c) {
  return __builtin_amdgcn_wmma_f32_16x16x32_bf16(false, a, false, b, (short)0, c,
                                                 false, false);
}

// ---------------------- TDM: async 128x32 bf16 tile -> LDS ----------------------
// D# per CDNA5 ISA (2-D tensor, groups 2/3 zero).

static __device__ __forceinline__ unsigned lds_addr_of(const void* p) {
  // flat LDS address: addr[31:0] is the LDS byte address
  return (unsigned)(uintptr_t)p;
}

static __device__ __forceinline__ void tdm_load_tile_128x32_bf16(
    unsigned lds_byte_addr, const __bf16* gsrc,
    unsigned rem_d0, unsigned rem_d1, unsigned stride_elems) {
  unsigned long long ga = (unsigned long long)(uintptr_t)gsrc;
  u32x4 g0;
  g0.x = 1u;                                        // count=1 (valid descriptor)
  g0.y = lds_byte_addr;                             // lds_addr (bytes)
  g0.z = (unsigned)(ga & 0xffffffffu);              // global_addr[31:0]
  g0.w = (unsigned)((ga >> 32) & 0x01ffffffu)       // global_addr[56:32]
       | (2u << 30);                                // type=2 ("image")
  i32x8 g1;
  g1[0] = 0x00010000;                               // data_size=1 (2B), mask=0
  g1[1] = (int)((rem_d0 & 0xffffu) << 16);          // tensor_dim0[15:0]
  g1[2] = (int)(((rem_d0 >> 16) & 0xffffu)          // tensor_dim0[31:16]
       | ((rem_d1 & 0xffffu) << 16));               // tensor_dim1[15:0]
  g1[3] = (int)(((rem_d1 >> 16) & 0xffffu)          // tensor_dim1[31:16]
       | (32u << 16));                              // tile_dim0 = 32 elems
  g1[4] = 128;                                      // tile_dim1 = 128 rows
  g1[5] = (int)stride_elems;                        // tensor_dim0_stride[31:0]
  g1[6] = 0;
  g1[7] = 0;
  i32x4 z4 = {0, 0, 0, 0};
#if defined(__clang_major__) && (__clang_major__ >= 23)
  i32x8 z8 = {};
  __builtin_amdgcn_tensor_load_to_lds(g0, g1, z4, z4, z8, 0);
#else
  __builtin_amdgcn_tensor_load_to_lds(g0, g1, z4, z4, 0);
#endif
}

// ---------------------------- precision conversion ----------------------------

__global__ __launch_bounds__(256) void convert_x(const float* __restrict__ x,
                                                 __bf16* __restrict__ y) {
  size_t i = ((size_t)blockIdx.x * blockDim.x + threadIdx.x) * 4;
  float4 v = *(const float4*)(x + i);
  bf16x4 o = { (__bf16)v.x, (__bf16)v.y, (__bf16)v.z, (__bf16)v.w };
  *(bf16x4*)(y + i) = o;
}

// transpose + convert: T[n*D + k] = (bf16) W[k*D + n]
__global__ __launch_bounds__(256) void wtrans(
    const float* __restrict__ W0, const float* __restrict__ W1,
    const float* __restrict__ W2, const float* __restrict__ W3,
    __bf16* __restrict__ T0, __bf16* __restrict__ T1,
    __bf16* __restrict__ T2, __bf16* __restrict__ T3) {
  const float* W = blockIdx.z == 0 ? W0 : blockIdx.z == 1 ? W1
                 : blockIdx.z == 2 ? W2 : W3;
  __bf16* T = blockIdx.z == 0 ? T0 : blockIdx.z == 1 ? T1
            : blockIdx.z == 2 ? T2 : T3;
  __shared__ float tile[32][33];
  int x  = blockIdx.x * 32 + threadIdx.x;
  int y0 = blockIdx.y * 32 + threadIdx.y;
#pragma unroll
  for (int i = 0; i < 4; ++i)
    tile[threadIdx.y + i * 8][threadIdx.x] = W[(size_t)(y0 + i * 8) * D_MODEL + x];
  __syncthreads();
  int tx  = blockIdx.y * 32 + threadIdx.x;
  int ty0 = blockIdx.x * 32 + threadIdx.y;
#pragma unroll
  for (int i = 0; i < 4; ++i)
    T[(size_t)(ty0 + i * 8) * D_MODEL + tx] =
        (__bf16)tile[threadIdx.x][threadIdx.y + i * 8];
}

// ------------------------------- QKV projection -------------------------------
// grid: (ROWS/128, D_MODEL/128, 3)  block: 256 (8 waves, each 32x64 C tile)
// A tile (128x32) staged in LDS by TDM, double buffered; weights direct from L2.
// z==0 -> Q (B,H,T,hd) ; z==1 -> K (B,H,T,hd) ; z==2 -> V transposed (B,H,hd,T)

__global__ __launch_bounds__(256) void qkv_gemm(
    const __bf16* __restrict__ Xb,
    const __bf16* __restrict__ Wqt, const __bf16* __restrict__ Wkt,
    const __bf16* __restrict__ Wvt,
    __bf16* __restrict__ Qb, __bf16* __restrict__ Kb, __bf16* __restrict__ Vt) {
  __shared__ __bf16 atile[2][128 * 32];
  const int wid = threadIdx.x >> 5;
  const int l = threadIdx.x & 31;
  const int g = l >> 4, n = l & 15;
  const int which = blockIdx.z;
  const __bf16* Wt = which == 0 ? Wqt : (which == 1 ? Wkt : Wvt);
  const int rowBase0 = blockIdx.x * 128;           // workgroup A-tile rows
  const int lr = (wid & 3) * 32;                   // wave rows inside A tile
  const int rowBase = rowBase0 + lr;
  const int colBase = blockIdx.y * 128 + (wid >> 2) * 64;

  if (wid == 0)
    tdm_load_tile_128x32_bf16(lds_addr_of(&atile[0][0]),
                              Xb + (size_t)rowBase0 * D_MODEL,
                              D_MODEL, ROWS - rowBase0, D_MODEL);

  f32x8 acc[2][4] = {};
  int buf = 0;
  for (int k0 = 0; k0 < D_MODEL; k0 += 32, buf ^= 1) {
    if (wid == 0) __builtin_amdgcn_s_wait_tensorcnt(0);
    __syncthreads();
    if (wid == 0 && k0 + 32 < D_MODEL)
      tdm_load_tile_128x32_bf16(lds_addr_of(&atile[buf ^ 1][0]),
                                Xb + (size_t)rowBase0 * D_MODEL + k0 + 32,
                                D_MODEL - (k0 + 32), ROWS - rowBase0, D_MODEL);
    const __bf16* at = &atile[buf][0];
    bf16x16 a0 = load_a_frag(at + (lr + 0) * 32, 32);
    bf16x16 a1 = load_a_frag(at + (lr + 16) * 32, 32);
#pragma unroll
    for (int ct = 0; ct < 4; ++ct) {
      bf16x16 bb = load_b_frag(Wt + (size_t)(colBase + ct * 16) * D_MODEL + k0, D_MODEL);
      acc[0][ct] = wmma_bf16(a0, bb, acc[0][ct]);
      acc[1][ct] = wmma_bf16(a1, bb, acc[1][ct]);
    }
    __syncthreads();
  }
#pragma unroll
  for (int mt = 0; mt < 2; ++mt)
#pragma unroll
    for (int ct = 0; ct < 4; ++ct)
#pragma unroll
      for (int r = 0; r < 8; ++r) {
        int row = rowBase + mt * 16 + g * 8 + r;
        int col = colBase + ct * 16 + n;
        int bb_ = row >> 11, t = row & (SEQ - 1);
        int hh = col >> 6, d = col & (HD - 1);
        __bf16 v = (__bf16)acc[mt][ct][r];
        if (which == 0)
          Qb[(((size_t)bb_ * NH + hh) * SEQ + t) * HD + d] = v;
        else if (which == 1)
          Kb[(((size_t)bb_ * NH + hh) * SEQ + t) * HD + d] = v;
        else
          Vt[(((size_t)bb_ * NH + hh) * HD + d) * SEQ + t] = v;
      }
}

// ----------------------------- flash attention --------------------------------
// grid: (SEQ/128, NH, BATCH)  block: 256; each wave owns 16 query rows.

__global__ __launch_bounds__(256) void attn_kernel(
    const __bf16* __restrict__ Qb, const __bf16* __restrict__ Kb,
    const __bf16* __restrict__ Vt, __bf16* __restrict__ Ob) {
  __shared__ __bf16 plds[8][16 * 32];
  const int wid = threadIdx.x >> 5;
  const int l = threadIdx.x & 31;
  const int g = l >> 4, n = l & 15;
  const int h = blockIdx.y, b = blockIdx.z;
  const int bh = b * NH + h;
  const int qBase = blockIdx.x * 128 + wid * 16;

  const __bf16* Qp = Qb + (size_t)bh * SEQ * HD;
  const __bf16* Kp = Kb + (size_t)bh * SEQ * HD;
  const __bf16* Vp = Vt + (size_t)bh * HD * SEQ;

  bf16x16 aq0 = load_a_frag(Qp + (size_t)qBase * HD, HD);
  bf16x16 aq1 = load_a_frag(Qp + (size_t)qBase * HD + 32, HD);

  f32x8 mi, li, o0, o1, o2, o3;
#pragma unroll
  for (int r = 0; r < 8; ++r) {
    mi[r] = -__builtin_inff(); li[r] = 0.f;
    o0[r] = o1[r] = o2[r] = o3[r] = 0.f;
  }
  const float sl = 0.125f * 1.44269504088896340736f;  // 1/sqrt(64) * log2(e)
  __bf16* pw = &plds[wid][0];

  for (int j0 = 0; j0 < qBase + 16; j0 += 32) {
    f32x8 s0{}, s1{};
    {
      bf16x16 b00 = load_b_frag(Kp + (size_t)j0 * HD, HD);
      bf16x16 b01 = load_b_frag(Kp + (size_t)j0 * HD + 32, HD);
      bf16x16 b10 = load_b_frag(Kp + (size_t)(j0 + 16) * HD, HD);
      bf16x16 b11 = load_b_frag(Kp + (size_t)(j0 + 16) * HD + 32, HD);
      s0 = wmma_bf16(aq0, b00, s0);
      s0 = wmma_bf16(aq1, b01, s0);
      s1 = wmma_bf16(aq0, b10, s1);
      s1 = wmma_bf16(aq1, b11, s1);
    }
    if (j0 + 32 < qBase + 16)  // global_prefetch_b8 next K block
      __builtin_prefetch(Kp + (size_t)(j0 + 32) * HD + l * 8, 0, 1);

    f32x8 p0, p1;
#pragma unroll
    for (int r = 0; r < 8; ++r) {
      int q = qBase + g * 8 + r;
      float v0 = (j0 + n      <= q) ? s0[r] * sl : -__builtin_inff();
      float v1 = (j0 + 16 + n <= q) ? s1[r] * sl : -__builtin_inff();
      float mrow = fmaxf(v0, v1);
      mrow = fmaxf(mrow, __shfl_xor(mrow, 1, 32));
      mrow = fmaxf(mrow, __shfl_xor(mrow, 2, 32));
      mrow = fmaxf(mrow, __shfl_xor(mrow, 4, 32));
      mrow = fmaxf(mrow, __shfl_xor(mrow, 8, 32));
      float mnew = fmaxf(mi[r], mrow);
      float cf = exp2f(mi[r] - mnew);
      float e0 = exp2f(v0 - mnew);
      float e1 = exp2f(v1 - mnew);
      float rs = e0 + e1;
      rs += __shfl_xor(rs, 1, 32);
      rs += __shfl_xor(rs, 2, 32);
      rs += __shfl_xor(rs, 4, 32);
      rs += __shfl_xor(rs, 8, 32);
      li[r] = li[r] * cf + rs;
      mi[r] = mnew;
      o0[r] *= cf; o1[r] *= cf; o2[r] *= cf; o3[r] *= cf;
      p0[r] = e0; p1[r] = e1;
    }

    // C-layout P -> LDS row-major 16x32 -> reload as A fragment (wave-local)
#pragma unroll
    for (int r = 0; r < 8; ++r) {
      int m = g * 8 + r;
      pw[m * 32 + n]      = (__bf16)p0[r];
      pw[m * 32 + 16 + n] = (__bf16)p1[r];
    }
    __builtin_amdgcn_wave_barrier();
    bf16x16 pa = load_a_frag(pw, 32);
    __builtin_amdgcn_wave_barrier();

    o0 = wmma_bf16(pa, load_b_frag(Vp + 0 * 16 * SEQ + j0, SEQ), o0);
    o1 = wmma_bf16(pa, load_b_frag(Vp + 1 * 16 * SEQ + j0, SEQ), o1);
    o2 = wmma_bf16(pa, load_b_frag(Vp + 2 * 16 * SEQ + j0, SEQ), o2);
    o3 = wmma_bf16(pa, load_b_frag(Vp + 3 * 16 * SEQ + j0, SEQ), o3);
  }

#pragma unroll
  for (int r = 0; r < 8; ++r) {
    float inv = 1.0f / li[r];
    int q = qBase + g * 8 + r;
    size_t rowoff = ((size_t)b * SEQ + q) * D_MODEL + h * HD;
    Ob[rowoff + 0 * 16 + n] = (__bf16)(o0[r] * inv);
    Ob[rowoff + 1 * 16 + n] = (__bf16)(o1[r] * inv);
    Ob[rowoff + 2 * 16 + n] = (__bf16)(o2[r] * inv);
    Ob[rowoff + 3 * 16 + n] = (__bf16)(o3[r] * inv);
  }
}

// ------------------------------ output projection ------------------------------

__global__ __launch_bounds__(256) void out_proj(
    const __bf16* __restrict__ Ab, const __bf16* __restrict__ Wot,
    const float* __restrict__ bo, float* __restrict__ out) {
  __shared__ __bf16 atile[2][128 * 32];
  const int wid = threadIdx.x >> 5;
  const int l = threadIdx.x & 31;
  const int g = l >> 4, n = l & 15;
  const int rowBase0 = blockIdx.x * 128;
  const int lr = (wid & 3) * 32;
  const int rowBase = rowBase0 + lr;
  const int colBase = blockIdx.y * 128 + (wid >> 2) * 64;

  if (wid == 0)
    tdm_load_tile_128x32_bf16(lds_addr_of(&atile[0][0]),
                              Ab + (size_t)rowBase0 * D_MODEL,
                              D_MODEL, ROWS - rowBase0, D_MODEL);

  f32x8 acc[2][4] = {};
  int buf = 0;
  for (int k0 = 0; k0 < D_MODEL; k0 += 32, buf ^= 1) {
    if (wid == 0) __builtin_amdgcn_s_wait_tensorcnt(0);
    __syncthreads();
    if (wid == 0 && k0 + 32 < D_MODEL)
      tdm_load_tile_128x32_bf16(lds_addr_of(&atile[buf ^ 1][0]),
                                Ab + (size_t)rowBase0 * D_MODEL + k0 + 32,
                                D_MODEL - (k0 + 32), ROWS - rowBase0, D_MODEL);
    const __bf16* at = &atile[buf][0];
    bf16x16 a0 = load_a_frag(at + (lr + 0) * 32, 32);
    bf16x16 a1 = load_a_frag(at + (lr + 16) * 32, 32);
#pragma unroll
    for (int ct = 0; ct < 4; ++ct) {
      bf16x16 bb = load_b_frag(Wot + (size_t)(colBase + ct * 16) * D_MODEL + k0, D_MODEL);
      acc[0][ct] = wmma_bf16(a0, bb, acc[0][ct]);
      acc[1][ct] = wmma_bf16(a1, bb, acc[1][ct]);
    }
    __syncthreads();
  }
#pragma unroll
  for (int ct = 0; ct < 4; ++ct) {
    int col = colBase + ct * 16 + n;
    float bias = bo[col];
#pragma unroll
    for (int mt = 0; mt < 2; ++mt)
#pragma unroll
      for (int r = 0; r < 8; ++r) {
        int row = rowBase + mt * 16 + g * 8 + r;
        out[(size_t)row * D_MODEL + col] = acc[mt][ct][r] + bias;
      }
  }
}

// ---------------------------------- launcher ----------------------------------

extern "C" void kernel_launch(void* const* d_in, const int* in_sizes, int n_in,
                              void* d_out, int out_size, void* d_ws, size_t ws_size,
                              hipStream_t stream) {
  const float* X  = (const float*)d_in[0];
  const float* Wq = (const float*)d_in[1];
  const float* Wk = (const float*)d_in[2];
  const float* Wv = (const float*)d_in[3];
  const float* Wo = (const float*)d_in[4];
  const float* bo = (const float*)d_in[5];
  float* out = (float*)d_out;

  char* p = (char*)d_ws;
  __bf16* Xb  = (__bf16*)p; p += (size_t)ROWS * D_MODEL * 2;
  __bf16* Wqt = (__bf16*)p; p += (size_t)D_MODEL * D_MODEL * 2;
  __bf16* Wkt = (__bf16*)p; p += (size_t)D_MODEL * D_MODEL * 2;
  __bf16* Wvt = (__bf16*)p; p += (size_t)D_MODEL * D_MODEL * 2;
  __bf16* Wot = (__bf16*)p; p += (size_t)D_MODEL * D_MODEL * 2;
  __bf16* Qb  = (__bf16*)p; p += (size_t)ROWS * D_MODEL * 2;
  __bf16* Kb  = (__bf16*)p; p += (size_t)ROWS * D_MODEL * 2;
  __bf16* Vt  = (__bf16*)p; p += (size_t)ROWS * D_MODEL * 2;
  __bf16* Ob  = (__bf16*)p; p += (size_t)ROWS * D_MODEL * 2;

  convert_x<<<(ROWS * D_MODEL) / (256 * 4), 256, 0, stream>>>(X, Xb);
  wtrans<<<dim3(32, 32, 4), dim3(32, 8), 0, stream>>>(Wq, Wk, Wv, Wo,
                                                      Wqt, Wkt, Wvt, Wot);
  qkv_gemm<<<dim3(ROWS / 128, D_MODEL / 128, 3), 256, 0, stream>>>(
      Xb, Wqt, Wkt, Wvt, Qb, Kb, Vt);
  attn_kernel<<<dim3(SEQ / 128, NH, BATCH), 256, 0, stream>>>(Qb, Kb, Vt, Ob);
  out_proj<<<dim3(ROWS / 128, D_MODEL / 128), 256, 0, stream>>>(Ob, Wot, bo, out);
}
